// AttnRNNDecoder_9740985827862
// MI455X (gfx1250) — matrix operand order repaired and checked by reading, not compiled, not run on hardware
//
#include <hip/hip_runtime.h>
#include <cstdint>
#include <cstddef>

// ---------------------------------------------------------------------------
// AttnRNNDecoder for MI455X (gfx1250, wave32, WMMA bf16 -> f32 accumulate)
// Register-blocked WMMA GEMM: each wave owns an RMx RN grid of 16x16 tiles.
// ---------------------------------------------------------------------------

typedef __attribute__((ext_vector_type(16))) __bf16 v16bf;
typedef __attribute__((ext_vector_type(8)))  __bf16 v8bf;
typedef __attribute__((ext_vector_type(8)))  float  v8f;

#define B_   32
#define T_   32
#define S_   64
#define H_   1024
#define E_   512
#define V_   32000
#define G4H  4096          // 4*H
#define LDI0 1537          // E + 1 + H (W_ih0 row length)

__device__ __forceinline__ float sigf(float x) { return 1.f / (1.f + expf(-x)); }

// ---------------------------------------------------------------------------
// C[M,N] = act( A[M,K](bf16) * Bw[N,K](bf16)^T + bias + Cadd )
// 4 waves per block along N; each wave computes a (16*RM) x (16*RN) C tile.
// Requires: M % (16*RM) == 0, N % (64*RN) == 0, K % 32 == 0.
// ---------------------------------------------------------------------------
template <int RM, int RN>
__global__ __launch_bounds__(128)
void gemm_bf16_nt(const __bf16* __restrict__ A, const __bf16* __restrict__ Bw,
                  const float* __restrict__ bias, const float* __restrict__ Cadd,
                  float* __restrict__ C, int M, int N, int K, int act)
{
  const int wave = threadIdx.x >> 5;
  const int lane = threadIdx.x & 31;
  const int tn0 = (blockIdx.x * 4 + wave) * (16 * RN);
  const int tm0 = blockIdx.y * (16 * RM);
  const int kg = lane >> 4;          // K-group select (0 or 1) per ISA layout
  const int lr = lane & 15;

  // A 16x32 bf16 layout: lanes 0-15 hold K {0..7,16..23}; lanes 16-31 K {8..15,24..31}
  const __bf16* ap[RM];
#pragma unroll
  for (int i = 0; i < RM; ++i)
    ap[i] = A + (size_t)(tm0 + 16 * i + lr) * K + kg * 8;
  // B 32x16 bf16 layout: lanes 0-15 hold K {0..7,16..23}; lanes 16-31 K {8..15,24..31}
  const __bf16* bp[RN];
#pragma unroll
  for (int j = 0; j < RN; ++j)
    bp[j] = Bw + (size_t)(tn0 + 16 * j + lr) * K + kg * 8;

  v8f acc[RM][RN] = {};

  for (int k0 = 0; k0 < K; k0 += 32) {
    v16bf av[RM], bv[RN];
#pragma unroll
    for (int i = 0; i < RM; ++i) {
      v8bf a0 = *(const v8bf*)(ap[i] + k0);
      v8bf a1 = *(const v8bf*)(ap[i] + k0 + 16);
      av[i] = __builtin_shufflevector(a0, a1, 0,1,2,3,4,5,6,7,8,9,10,11,12,13,14,15);
    }
#pragma unroll
    for (int j = 0; j < RN; ++j) {
      v8bf b0 = *(const v8bf*)(bp[j] + k0);
      v8bf b1 = *(const v8bf*)(bp[j] + k0 + 16);
      bv[j] = __builtin_shufflevector(b0, b1, 0,1,2,3,4,5,6,7,8,9,10,11,12,13,14,15);
    }
#pragma unroll
    for (int i = 0; i < RM; ++i)
#pragma unroll
      for (int j = 0; j < RN; ++j)
        acc[i][j] = __builtin_amdgcn_wmma_f32_16x16x32_bf16(
            false, av[i], false, bv[j], (short)0, acc[i][j], false, false);
  }

  // C/D layout: lane -> N (=lr), VGPR v -> M = v + 8*(lane>=16)
#pragma unroll
  for (int j = 0; j < RN; ++j) {
    const int n = tn0 + 16 * j + lr;
    const float bn = bias ? bias[n] : 0.f;
#pragma unroll
    for (int i = 0; i < RM; ++i) {
#pragma unroll
      for (int v = 0; v < 8; ++v) {
        const int m = tm0 + 16 * i + v + 8 * kg;
        const size_t idx = (size_t)m * N + n;
        float x = acc[i][j][v] + bn;
        if (Cadd) x += Cadd[idx];
        if (act)  x = tanhf(x);
        C[idx] = x;
      }
    }
  }
}

// ---------------------------------------------------------------------------
// Weight prep kernels (f32 -> bf16, with concat packing for fused GEMMs)
// ---------------------------------------------------------------------------
__global__ void f2bf(const float* __restrict__ s, __bf16* __restrict__ d, size_t n)
{
  for (size_t i = (size_t)blockIdx.x * blockDim.x + threadIdx.x; i < n;
       i += (size_t)gridDim.x * blockDim.x)
    d[i] = (__bf16)s[i];
}

__global__ void f2bf_strided(const float* __restrict__ s, int lds, int offs,
                             __bf16* __restrict__ d, int N, int K)
{
  size_t tot = (size_t)N * K;
  for (size_t i = (size_t)blockIdx.x * blockDim.x + threadIdx.x; i < tot;
       i += (size_t)gridDim.x * blockDim.x) {
    int n = (int)(i / K), k = (int)(i % K);
    d[i] = (__bf16)s[(size_t)n * lds + offs + k];
  }
}

// dst[n, 0:Kh] = A[n, offA:offA+Kh];  dst[n, Kh:2Kh] = Bm[n, 0:Kh]
__global__ void pack_cat(const float* __restrict__ A, int ldA, int offA,
                         const float* __restrict__ Bm, int ldB,
                         __bf16* __restrict__ dst, int N, int Kh)
{
  size_t tot = (size_t)N * 2 * Kh;
  for (size_t i = (size_t)blockIdx.x * blockDim.x + threadIdx.x; i < tot;
       i += (size_t)gridDim.x * blockDim.x) {
    int n = (int)(i / (2 * Kh)), k = (int)(i % (2 * Kh));
    float v = (k < Kh) ? A[(size_t)n * ldA + offA + k]
                       : Bm[(size_t)n * ldB + (k - Kh)];
    dst[i] = (__bf16)v;
  }
}

__global__ void misc_prep(const float* __restrict__ Wih0,
                          const float* __restrict__ bi0, const float* __restrict__ bh0,
                          const float* __restrict__ bi1, const float* __restrict__ bh1,
                          float* __restrict__ wlc, float* __restrict__ bias0,
                          float* __restrict__ bias1)
{
  int n = blockIdx.x * blockDim.x + threadIdx.x;
  if (n >= G4H) return;
  wlc[n]   = Wih0[(size_t)n * LDI0 + E_];   // countdown column of W_ih0
  bias0[n] = bi0[n] + bh0[n];
  bias1[n] = bi1[n] + bh1[n];
}

// gather embeddings (to bf16) for all timesteps; also transpose lc to [T,B]
__global__ void embed_kernel(const int* __restrict__ ids, const float* __restrict__ lc,
                             const float* __restrict__ E, __bf16* __restrict__ Embg,
                             float* __restrict__ lcv)
{
  size_t i = (size_t)blockIdx.x * blockDim.x + threadIdx.x;   // exact: T*B*E
  int m = (int)(i / E_), k = (int)(i % E_);
  int t = m / B_, b = m % B_;
  Embg[i] = (__bf16)E[(size_t)ids[b * T_ + t] * E_ + k];
  if (k == 0) lcv[m] = lc[b * T_ + t];
}

// Gpre[m,n] += lcv[m] * wlc[n]   (rank-1 countdown term)
__global__ void add_lc(float* __restrict__ Gpre, const float* __restrict__ lcv,
                       const float* __restrict__ wlc)
{
  size_t i = (size_t)blockIdx.x * blockDim.x + threadIdx.x;   // exact: T*B*4H
  int m = (int)(i / G4H), n = (int)(i % G4H);
  Gpre[i] += lcv[m] * wlc[n];
}

// ---------------------------------------------------------------------------
// State init / LSTM pointwise / attention / epilogue kernels
// ---------------------------------------------------------------------------
__global__ void init_state(const float* __restrict__ h0, const float* __restrict__ c0,
                           const float* __restrict__ prev, float* __restrict__ h,
                           float* __restrict__ c, __bf16* __restrict__ xcat0,
                           __bf16* __restrict__ xcat1)
{
  int idx = blockIdx.x * blockDim.x + threadIdx.x;   // exact: B*H
  int b = idx >> 10, j = idx & (H_ - 1);
  float h0v = h0[idx], h1v = h0[B_ * H_ + idx];
  h[idx] = h0v;             h[B_ * H_ + idx] = h1v;
  c[idx] = c0[idx];         c[B_ * H_ + idx] = c0[B_ * H_ + idx];
  xcat0[(size_t)b * 2 * H_ + j]       = (__bf16)prev[idx];  // input feed
  xcat0[(size_t)b * 2 * H_ + H_ + j]  = (__bf16)h0v;        // h layer 0
  xcat1[(size_t)b * 2 * H_ + H_ + j]  = (__bf16)h1v;        // h layer 1
}

// gates G[B,4H] (order i,f,g,o) + c -> new c,h ; fan h out to bf16 buffers
__global__ __launch_bounds__(256)
void lstm_cell(const float* __restrict__ G, float* __restrict__ c, float* __restrict__ h,
               __bf16* d0, int ld0, int off0,
               __bf16* d1, int ld1, int off1,
               __bf16* d2, int ld2, int off2)
{
  int idx = blockIdx.x * blockDim.x + threadIdx.x;   // exact: B*H
  int b = idx >> 10, j = idx & (H_ - 1);
  const float* g = G + (size_t)b * G4H;
  float iv = g[j], fv = g[H_ + j], gv = g[2 * H_ + j], ov = g[3 * H_ + j];
  float cn = sigf(fv) * c[idx] + sigf(iv) * tanhf(gv);
  float hn = sigf(ov) * tanhf(cn);
  c[idx] = cn;
  h[idx] = hn;
  __bf16 hb = (__bf16)hn;
  if (d0) d0[(size_t)b * ld0 + off0 + j] = hb;
  if (d1) d1[(size_t)b * ld1 + off1 + j] = hb;
  if (d2) d2[(size_t)b * ld2 + off2 + j] = hb;
}

// scores over S=64 + softmax + weighted context; writes wc into attin[:, 0:H]
__global__ __launch_bounds__(64)
void attn_kernel(const float* __restrict__ ctx,            // [S,B,H]
                 const unsigned char* __restrict__ mask,   // [B,S]
                 const float* __restrict__ q,              // [B,H]
                 __bf16* __restrict__ attin)               // [B,2H] -> first H
{
  __shared__ float sm[S_];
  __shared__ float red[S_];
  int b = blockIdx.x;
  int s = threadIdx.x;

  const float* cr = ctx + ((size_t)s * B_ + b) * H_;
  const float* qr = q + (size_t)b * H_;
  float a = 0.f;
  for (int h2 = 0; h2 < H_; ++h2) a += cr[h2] * qr[h2];
  if (!mask[b * S_ + s]) a = -1e30f;

  red[s] = a; __syncthreads();
  for (int st = 32; st > 0; st >>= 1) { if (s < st) red[s] = fmaxf(red[s], red[s + st]); __syncthreads(); }
  float mx = red[0]; __syncthreads();
  float e = expf(a - mx);
  red[s] = e; __syncthreads();
  for (int st = 32; st > 0; st >>= 1) { if (s < st) red[s] += red[s + st]; __syncthreads(); }
  float inv = 1.f / red[0]; __syncthreads();
  sm[s] = e * inv; __syncthreads();

  for (int h2 = s; h2 < H_; h2 += S_) {
    float acc = 0.f;
    for (int ss = 0; ss < S_; ++ss) acc += sm[ss] * ctx[((size_t)ss * B_ + b) * H_ + h2];
    attin[(size_t)b * 2 * H_ + h2] = (__bf16)acc;
  }
}

// out_cur (f32) -> OutAll[t] bf16 (for vocab GEMM) and xcat0[:,0:H] (next-step feed)
__global__ void post_att(const float* __restrict__ out, __bf16* __restrict__ outall,
                         __bf16* __restrict__ xcat0)
{
  int idx = blockIdx.x * blockDim.x + threadIdx.x;   // exact: B*H
  int b = idx >> 10, j = idx & (H_ - 1);
  __bf16 v = (__bf16)out[idx];
  outall[idx] = v;
  xcat0[(size_t)b * 2 * H_ + j] = v;
}

// in-place row log-softmax over V=32000
__global__ __launch_bounds__(256)
void logsoftmax_kernel(float* __restrict__ X)
{
  __shared__ float red[256];
  float* x = X + (size_t)blockIdx.x * V_;
  int tid = threadIdx.x;
  float m = -3.4e38f;
  for (int i = tid; i < V_; i += 256) m = fmaxf(m, x[i]);
  red[tid] = m; __syncthreads();
  for (int s = 128; s > 0; s >>= 1) { if (tid < s) red[tid] = fmaxf(red[tid], red[tid + s]); __syncthreads(); }
  m = red[0]; __syncthreads();
  float sum = 0.f;
  for (int i = tid; i < V_; i += 256) sum += expf(x[i] - m);
  red[tid] = sum; __syncthreads();
  for (int s = 128; s > 0; s >>= 1) { if (tid < s) red[tid] += red[tid + s]; __syncthreads(); }
  float lse = logf(red[0]) + m;
  for (int i = tid; i < V_; i += 256) x[i] = x[i] - lse;
}

__global__ void copy_states(const float* __restrict__ h, const float* __restrict__ c,
                            const float* __restrict__ o, float* __restrict__ hT,
                            float* __restrict__ cT, float* __restrict__ oT)
{
  int idx = blockIdx.x * blockDim.x + threadIdx.x;   // exact: 2*B*H
  hT[idx] = h[idx];
  cT[idx] = c[idx];
  if (idx < B_ * H_) oT[idx] = o[idx];
}

// ---------------------------------------------------------------------------
// Host side
// ---------------------------------------------------------------------------
extern "C" void kernel_launch(void* const* d_in, const int* in_sizes, int n_in,
                              void* d_out, int out_size, void* d_ws, size_t ws_size,
                              hipStream_t stream)
{
  const int*   ids    = (const int*)d_in[0];
  const float* lc     = (const float*)d_in[1];
  const float* h0in   = (const float*)d_in[2];
  const float* c0in   = (const float*)d_in[3];
  const float* ctx    = (const float*)d_in[4];
  const unsigned char* mask = (const unsigned char*)d_in[5];
  const float* prev   = (const float*)d_in[6];
  const float* E      = (const float*)d_in[7];
  const float* Wih0   = (const float*)d_in[8];
  const float* Whh0   = (const float*)d_in[9];
  const float* bih0   = (const float*)d_in[10];
  const float* bhh0   = (const float*)d_in[11];
  const float* Wih1   = (const float*)d_in[12];
  const float* Whh1   = (const float*)d_in[13];
  const float* bih1   = (const float*)d_in[14];
  const float* bhh1   = (const float*)d_in[15];
  const float* Walign = (const float*)d_in[16];
  const float* Wctx   = (const float*)d_in[17];
  const float* Wquery = (const float*)d_in[18];
  const float* Wout   = (const float*)d_in[19];
  const float* bout   = (const float*)d_in[20];

  // bump allocator over d_ws (256B aligned); total ~131 MB
  char* p = (char*)d_ws;
  auto alloc = [&](size_t bytes) -> void* {
    void* r = (void*)p;
    p += (bytes + 255) & ~(size_t)255;
    return r;
  };

  __bf16* Wemb_bf  = (__bf16*)alloc((size_t)G4H * E_ * 2);
  __bf16* Wcat0_bf = (__bf16*)alloc((size_t)G4H * 2 * H_ * 2);  // [W_feed | W_hh0]
  __bf16* Wcat1_bf = (__bf16*)alloc((size_t)G4H * 2 * H_ * 2);  // [W_ih1  | W_hh1]
  __bf16* Wcatt_bf = (__bf16*)alloc((size_t)H_  * 2 * H_ * 2);  // [W_ctx  | W_query]
  __bf16* Wal_bf   = (__bf16*)alloc((size_t)H_  * H_ * 2);
  __bf16* Wout_bf  = (__bf16*)alloc((size_t)V_  * H_ * 2);
  float*  wlc      = (float*)alloc(G4H * 4);
  float*  bias0    = (float*)alloc(G4H * 4);
  float*  bias1    = (float*)alloc(G4H * 4);
  __bf16* Embg_bf  = (__bf16*)alloc((size_t)T_ * B_ * E_ * 2);
  float*  lcv      = (float*)alloc(T_ * B_ * 4);
  float*  Gpre     = (float*)alloc((size_t)T_ * B_ * G4H * 4);
  float*  G0       = (float*)alloc((size_t)B_ * G4H * 4);
  float*  G1       = (float*)alloc((size_t)B_ * G4H * 4);
  float*  h_cur    = (float*)alloc(2 * B_ * H_ * 4);
  float*  c_cur    = (float*)alloc(2 * B_ * H_ * 4);
  float*  out_cur  = (float*)alloc(B_ * H_ * 4);
  __bf16* xcat0    = (__bf16*)alloc((size_t)B_ * 2 * H_ * 2);   // [out_prev | h0_prev]
  __bf16* xcat1    = (__bf16*)alloc((size_t)B_ * 2 * H_ * 2);   // [h0_new   | h1_prev]
  __bf16* attin    = (__bf16*)alloc((size_t)B_ * 2 * H_ * 2);   // [wc       | h1_new ]
  __bf16* h1bf     = (__bf16*)alloc((size_t)B_ * H_ * 2);
  float*  q        = (float*)alloc(B_ * H_ * 4);
  __bf16* OutAll   = (__bf16*)alloc((size_t)T_ * B_ * H_ * 2);

  // ---- one-time (per launch) weight conversion / packing --------------------
  f2bf_strided<<<2048, 256, 0, stream>>>(Wih0, LDI0, 0, Wemb_bf, G4H, E_);
  pack_cat<<<4096, 256, 0, stream>>>(Wih0, LDI0, E_ + 1, Whh0, H_, Wcat0_bf, G4H, H_);
  pack_cat<<<4096, 256, 0, stream>>>(Wih1, H_, 0, Whh1, H_, Wcat1_bf, G4H, H_);
  pack_cat<<<2048, 256, 0, stream>>>(Wctx, H_, 0, Wquery, H_, Wcatt_bf, H_, H_);
  f2bf<<<2048, 256, 0, stream>>>(Walign, Wal_bf, (size_t)H_ * H_);
  f2bf<<<8192, 256, 0, stream>>>(Wout, Wout_bf, (size_t)V_ * H_);
  misc_prep<<<16, 256, 0, stream>>>(Wih0, bih0, bhh0, bih1, bhh1, wlc, bias0, bias1);

  // ---- batched pre-scan: Gpre = emb @ Wemb^T + bias0 + lc*wlc ---------------
  embed_kernel<<<2048, 256, 0, stream>>>(ids, lc, E, Embg_bf, lcv);
  {
    dim3 g(G4H / 256, (T_ * B_) / 64);   // RM=4, RN=4
    gemm_bf16_nt<4, 4><<<g, 128, 0, stream>>>(Embg_bf, Wemb_bf, bias0, nullptr, Gpre,
                                              T_ * B_, G4H, E_, 0);
  }
  add_lc<<<16384, 256, 0, stream>>>(Gpre, lcv, wlc);
  init_state<<<128, 256, 0, stream>>>(h0in, c0in, prev, h_cur, c_cur, xcat0, xcat1);

  // ---- sequential scan over T (recurrent weights are L2-resident) -----------
  for (int t = 0; t < T_; ++t) {
    {  // layer-0 gates: Gpre[t] + [out,h0] @ [W_feed|W_hh0]^T
      dim3 g(G4H / 128, 1);              // RM=2, RN=2 (M=32)
      gemm_bf16_nt<2, 2><<<g, 128, 0, stream>>>(xcat0, Wcat0_bf, nullptr,
                                                Gpre + (size_t)t * B_ * G4H, G0,
                                                B_, G4H, 2 * H_, 0);
    }
    lstm_cell<<<128, 256, 0, stream>>>(G0, c_cur, h_cur,
                                       xcat1, 2 * H_, 0,          // h0_new -> xcat1 lo
                                       xcat0, 2 * H_, H_,         // h0_new -> xcat0 hi
                                       (__bf16*)nullptr, 0, 0);
    {  // layer-1 gates: bias1 + [h0_new,h1_prev] @ [W_ih1|W_hh1]^T
      dim3 g(G4H / 128, 1);
      gemm_bf16_nt<2, 2><<<g, 128, 0, stream>>>(xcat1, Wcat1_bf, bias1, nullptr, G1,
                                                B_, G4H, 2 * H_, 0);
    }
    lstm_cell<<<128, 256, 0, stream>>>(G1, c_cur + B_ * H_, h_cur + B_ * H_,
                                       h1bf, H_, 0,               // h1_new for q GEMM
                                       attin, 2 * H_, H_,         // h1_new -> attin hi
                                       xcat1, 2 * H_, H_);        // h1_new for next step
    {  // q = h1_new @ W_align^T
      dim3 g(H_ / 128, 1);
      gemm_bf16_nt<2, 2><<<g, 128, 0, stream>>>(h1bf, Wal_bf, nullptr, nullptr, q,
                                                B_, H_, H_, 0);
    }
    attn_kernel<<<B_, 64, 0, stream>>>(ctx, mask, q, attin);
    {  // out = tanh([wc,h1_new] @ [W_ctx|W_query]^T)
      dim3 g(H_ / 128, 1);
      gemm_bf16_nt<2, 2><<<g, 128, 0, stream>>>(attin, Wcatt_bf, nullptr, nullptr,
                                                out_cur, B_, H_, 2 * H_, 1);
    }
    post_att<<<128, 256, 0, stream>>>(out_cur, OutAll + (size_t)t * B_ * H_, xcat0);
  }

  // ---- batched post-scan: vocab projection + log-softmax, straight into d_out
  float* scores = (float*)d_out;                       // [T,B,V]
  float* hT = scores + (size_t)T_ * B_ * V_;           // [2,B,H]
  float* cT = hT + 2 * B_ * H_;                        // [2,B,H]
  float* oT = cT + 2 * B_ * H_;                        // [B,H]
  {
    dim3 g(V_ / 256, (T_ * B_) / 64);    // RM=4, RN=4
    gemm_bf16_nt<4, 4><<<g, 128, 0, stream>>>(OutAll, Wout_bf, bout, nullptr, scores,
                                              T_ * B_, V_, H_, 0);
  }
  logsoftmax_kernel<<<T_ * B_, 256, 0, stream>>>(scores);
  copy_states<<<256, 256, 0, stream>>>(h_cur, c_cur, out_cur, hT, cT, oT);
}
